// Encoder_30374008717735
// MI455X (gfx1250) — matrix operand order repaired and checked by reading, not compiled, not run on hardware
//
#include <hip/hip_runtime.h>

typedef __attribute__((ext_vector_type(16))) _Float16 v16h;
typedef __attribute__((ext_vector_type(8)))  float    v8f;

#define L_SEQ   128
#define B_BATCH 64
#define E_DIM   512
#define H_DIM   512
#define G_DIM   1536   // 3*H

static constexpr size_t WT_ELEMS = (size_t)G_DIM * E_DIM;            // f16 elems per transposed weight
static constexpr size_t XW_ELEMS = (size_t)L_SEQ * B_BATCH * G_DIM;  // f32 elems per direction

// ---------------------------------------------------------------------------
// Fast activations
// ---------------------------------------------------------------------------
__device__ __forceinline__ float sigf(float x) {
    return 1.0f / (1.0f + __expf(-x));
}
__device__ __forceinline__ float tanh_fast(float x) {
    return 2.0f / (1.0f + __expf(-2.0f * x)) - 1.0f;
}

// ---------------------------------------------------------------------------
// A fragment (16x32 f16, M=lane%16) loaded from a row-major f32 row pointer.
// Contiguous K order (matches B): lanes 0-15 hold K = k0+{0..15},
//                                 lanes 16-31 hold K = k0+{16..31}.
// ---------------------------------------------------------------------------
__device__ __forceinline__ v16h a_frag_f32row(const float* __restrict__ row,
                                              int k0, int half) {
    const int b1 = k0 + (half ? 16 : 0);
    const int b2 = b1 + 8;
    v16h a;
#pragma unroll
    for (int i = 0; i < 8; ++i) {
        a[i]     = (_Float16)row[b1 + i];
        a[8 + i] = (_Float16)row[b2 + i];
    }
    return a;
}

// ---------------------------------------------------------------------------
// Kernel 1: convert Wx_f, Wx_b, Wh_f, Wh_b (512x1536 f32, row-major) into
// f16 N-major (1536x512) so B fragments are contiguous 32B loads per lane.
// LDS tiled transpose, 32x32 tiles.
// ---------------------------------------------------------------------------
__global__ void __launch_bounds__(256)
prep_weights(const float* __restrict__ Wx_f, const float* __restrict__ Wx_b,
             const float* __restrict__ Wh_f, const float* __restrict__ Wh_b,
             _Float16* __restrict__ wt) {
    __shared__ float tile[32][33];
    const int w = blockIdx.z;
    const float* W = (w == 0) ? Wx_f : (w == 1) ? Wx_b : (w == 2) ? Wh_f : Wh_b;
    _Float16* Wt = wt + (size_t)w * WT_ELEMS;

    const int n0 = blockIdx.x * 32;   // column in W == row in Wt
    const int k0 = blockIdx.y * 32;   // row in W
#pragma unroll
    for (int i = 0; i < 4; ++i) {
        int k = k0 + threadIdx.y + 8 * i;
        tile[threadIdx.y + 8 * i][threadIdx.x] =
            W[(size_t)k * G_DIM + n0 + threadIdx.x];
    }
    __syncthreads();
#pragma unroll
    for (int i = 0; i < 4; ++i) {
        int n = n0 + threadIdx.y + 8 * i;
        Wt[(size_t)n * E_DIM + k0 + threadIdx.x] =
            (_Float16)tile[threadIdx.x][threadIdx.y + 8 * i];
    }
}

// ---------------------------------------------------------------------------
// Kernel 2: xw[dir] = emb[xs] @ Wx[dir] + b[dir]   (8192 x 1536, K=512)
// One wave computes a 16x64 output strip (4 WMMA tiles sharing the A frag).
// Block = 256 threads (8 waves). grid = (512 row tiles, 3, 2 directions).
// ---------------------------------------------------------------------------
__global__ void __launch_bounds__(256)
xw_gemm(const int* __restrict__ xs, const float* __restrict__ emb,
        const _Float16* __restrict__ wt_x_f, const _Float16* __restrict__ wt_x_b,
        const float* __restrict__ b_f, const float* __restrict__ b_b,
        float* __restrict__ xw_f, float* __restrict__ xw_b) {
    const int lane = threadIdx.x & 31, wave = threadIdx.x >> 5;
    const int half = lane >> 4, lm = lane & 15;
    const int dir  = blockIdx.z;

    const _Float16* Wt  = dir ? wt_x_b : wt_x_f;
    const float*   bias = dir ? b_b    : b_f;
    float*         out  = dir ? xw_b   : xw_f;

    const int m0   = blockIdx.x * 16;
    const int ncol = (blockIdx.y * 8 + wave) * 64;

    const int tok = xs[m0 + lm];                 // this lane's A row token
    const float* arow = emb + (size_t)tok * E_DIM;

    v8f c0 = {}, c1 = {}, c2 = {}, c3 = {};
    for (int k0 = 0; k0 < E_DIM; k0 += 32) {
        v16h a = a_frag_f32row(arow, k0, half);
        const _Float16* bp = Wt + (size_t)(ncol + lm) * E_DIM + k0 + half * 16;
        v16h bv0 = *(const v16h*)(bp);
        v16h bv1 = *(const v16h*)(bp + 16 * E_DIM);
        v16h bv2 = *(const v16h*)(bp + 32 * E_DIM);
        v16h bv3 = *(const v16h*)(bp + 48 * E_DIM);
        c0 = __builtin_amdgcn_wmma_f32_16x16x32_f16(false, a, false, bv0, (short)0, c0, false, false);
        c1 = __builtin_amdgcn_wmma_f32_16x16x32_f16(false, a, false, bv1, (short)0, c1, false, false);
        c2 = __builtin_amdgcn_wmma_f32_16x16x32_f16(false, a, false, bv2, (short)0, c2, false, false);
        c3 = __builtin_amdgcn_wmma_f32_16x16x32_f16(false, a, false, bv3, (short)0, c3, false, false);
    }
#pragma unroll
    for (int j = 0; j < 4; ++j) {
        const int n  = ncol + j * 16 + lm;
        const float bv = bias[n];
        v8f c = (j == 0) ? c0 : (j == 1) ? c1 : (j == 2) ? c2 : c3;
#pragma unroll
        for (int r = 0; r < 8; ++r) {
            const int m = m0 + r + half * 8;
            out[(size_t)m * G_DIM + n] = c[r] + bv;
        }
    }
}

// ---------------------------------------------------------------------------
// Kernel 3: one recurrence step, both directions at once.
//   forward  : t = t_idx,        h_prev = r1[t-1][:, 0:512]   (0 if t_idx==0)
//   backward : t = L-1-t_idx,    h_prev = r1[t+1][:, 512:1024](0 if t_idx==0)
// Each block handles one 16-wide hidden-column group for one direction.
// Wave w computes the 16-row batch tile for gates z/r/h (3 WMMA accumulators,
// K=512 -> 16 wmma iterations per gate), then applies the GRU gating and
// writes h_new straight into r1 (and r2 for backward t==0).
// grid = (32, 2), block = 128 (4 waves).
// ---------------------------------------------------------------------------
__global__ void __launch_bounds__(128)
gru_step(const float* __restrict__ xw_f, const float* __restrict__ xw_b,
         const _Float16* __restrict__ wt_h_f, const _Float16* __restrict__ wt_h_b,
         const float* __restrict__ mask, float* __restrict__ out, int t_idx) {
    const int lane = threadIdx.x & 31, wave = threadIdx.x >> 5;
    const int half = lane >> 4, lm = lane & 15;
    const int dir  = blockIdx.y;
    const int t    = dir ? (L_SEQ - 1 - t_idx) : t_idx;
    const bool first = (t_idx == 0);

    const _Float16* Wt = dir ? wt_h_b : wt_h_f;
    const float*    xw = dir ? xw_b   : xw_f;
    const int   colsel = dir ? H_DIM  : 0;     // which half of r1 rows

    const int m0 = wave * 16;                  // batch tile
    const int n0 = blockIdx.x * 16;            // hidden column tile

    const float* hprev = nullptr;
    if (!first) {
        const int tp = dir ? (t + 1) : (t - 1);
        hprev = out + (size_t)tp * B_BATCH * (2 * H_DIM) + colsel;
    }

    v8f cz = {}, cr = {}, ch = {};
    if (!first) {
        const float* arow = hprev + (size_t)(m0 + lm) * (2 * H_DIM);
        for (int k0 = 0; k0 < H_DIM; k0 += 32) {
            v16h a = a_frag_f32row(arow, k0, half);
            const _Float16* bp = Wt + (size_t)(n0 + lm) * H_DIM + k0 + half * 16;
            v16h bz = *(const v16h*)(bp);
            v16h br = *(const v16h*)(bp + (size_t)H_DIM * H_DIM);
            v16h bh = *(const v16h*)(bp + (size_t)(2 * H_DIM) * H_DIM);
            cz = __builtin_amdgcn_wmma_f32_16x16x32_f16(false, a, false, bz, (short)0, cz, false, false);
            cr = __builtin_amdgcn_wmma_f32_16x16x32_f16(false, a, false, br, (short)0, cr, false, false);
            ch = __builtin_amdgcn_wmma_f32_16x16x32_f16(false, a, false, bh, (short)0, ch, false, false);
        }
    }

    const float* xwt    = xw  + (size_t)t * B_BATCH * G_DIM;
    float*       outrow = out + (size_t)t * B_BATCH * (2 * H_DIM) + colsel;
    float*       r2     = out + (size_t)L_SEQ * B_BATCH * (2 * H_DIM);

    const int n = n0 + lm;
#pragma unroll
    for (int r = 0; r < 8; ++r) {
        const int mb = m0 + r + half * 8;
        const float* xrw = xwt + (size_t)mb * G_DIM;
        const float xz = xrw[n];
        const float xr = xrw[H_DIM + n];
        const float xh = xrw[2 * H_DIM + n];
        const float z  = sigf(xz + cz[r]);
        const float rr = sigf(xr + cr[r]);
        const float ht = tanh_fast(xh + rr * ch[r]);
        const float hp = first ? 0.0f : hprev[(size_t)mb * (2 * H_DIM) + n];
        float hn = (1.0f - z) * hp + z * ht;
        const float mv = mask[t * B_BATCH + mb];
        hn = mv * hn + (1.0f - mv) * hp;
        outrow[(size_t)mb * (2 * H_DIM) + n] = hn;
        if (dir && t == 0) r2[(size_t)mb * H_DIM + n] = hn;   // r2 = left[0]
    }
}

// ---------------------------------------------------------------------------
// Launch
// inputs: 0 xs(int L*B) 1 xs_mask 2 emb 3 Wx_f 4 Wh_f 5 b_f 6 Wx_b 7 Wh_b 8 b_b
// d_out: r1 (L,B,2H) f32 then r2 (B,H) f32
// ws: [Wx_f^T | Wx_b^T | Wh_f^T | Wh_b^T] f16, then xw_f, xw_b f32
// ---------------------------------------------------------------------------
extern "C" void kernel_launch(void* const* d_in, const int* in_sizes, int n_in,
                              void* d_out, int out_size, void* d_ws, size_t ws_size,
                              hipStream_t stream) {
    (void)in_sizes; (void)n_in; (void)out_size; (void)ws_size;
    const int*   xs   = (const int*)d_in[0];
    const float* mask = (const float*)d_in[1];
    const float* emb  = (const float*)d_in[2];
    const float* Wx_f = (const float*)d_in[3];
    const float* Wh_f = (const float*)d_in[4];
    const float* b_f  = (const float*)d_in[5];
    const float* Wx_b = (const float*)d_in[6];
    const float* Wh_b = (const float*)d_in[7];
    const float* b_b  = (const float*)d_in[8];
    float* out = (float*)d_out;

    _Float16* wt   = (_Float16*)d_ws;
    float*    xw_f = (float*)((char*)d_ws + 4 * WT_ELEMS * sizeof(_Float16));
    float*    xw_b = xw_f + XW_ELEMS;

    // 1) transpose + f16-convert all four weight matrices
    prep_weights<<<dim3(G_DIM / 32, E_DIM / 32, 4), dim3(32, 8), 0, stream>>>(
        Wx_f, Wx_b, Wh_f, Wh_b, wt);

    // 2) batched input projection for both directions
    xw_gemm<<<dim3((L_SEQ * B_BATCH) / 16, 3, 2), 256, 0, stream>>>(
        xs, emb, wt + 0 * WT_ELEMS, wt + 1 * WT_ELEMS, b_f, b_b, xw_f, xw_b);

    // 3) 128 stream-ordered recurrence steps (fwd t and bwd L-1-t per launch)
    const _Float16* wt_h_f = wt + 2 * WT_ELEMS;
    const _Float16* wt_h_b = wt + 3 * WT_ELEMS;
    for (int s = 0; s < L_SEQ; ++s) {
        gru_step<<<dim3(H_DIM / 16, 2), 128, 0, stream>>>(
            xw_f, xw_b, wt_h_f, wt_h_b, mask, out, s);
    }
}
